// HetGAT_24438363914895
// MI455X (gfx1250) — compile-verified, matrix operand order checked by reading, not run on hardware
//
#include <hip/hip_runtime.h>
#include <math.h>

typedef __attribute__((ext_vector_type(16))) __bf16 v16bf;
typedef __attribute__((ext_vector_type(8)))  float  v8f;

#define NROW 129          // 1 + NA + NO
#define MPAD 144          // 9 tiles of 16
#define KD   128          // D
#define KP   136          // padded LDS K stride (bank-conflict break)
#define NT   3            // T
#define CC   64           // C
#define NCOL (NT*CC)      // 192
#define HP   200          // padded H row stride (floats)

// LDS layout (byte offsets)
#define OFF_A    0                          // 144*136*2 = 39168
#define OFF_B    39168                      // 192*136*2 = 52224
#define OFF_H    91392                      // 144*200*4 = 115200
#define OFF_LOG  206592                     // 2*576*4   = 4608
#define OFF_ES   211200                     // 16*4      = 64
#define OFF_W    211264                     // 384*4     = 1536
#define OFF_RED  212800                     // 256*4     = 1024
#define OFF_PART 213824                     // 256*4     = 1024
#define OFF_AV   214848                     // 384*4     = 1536
#define SMEM_BYTES 216384

__device__ __forceinline__ __bf16 f2bf(float f) {
    union { float f; unsigned u; } x; x.f = f;
    unsigned short s = (unsigned short)(x.u >> 16);
    __bf16 r;
    __builtin_memcpy(&r, &s, 2);
    return r;
}

__global__ __launch_bounds__(256) void hetgat_fused(
    const float* __restrict__ h, const float* __restrict__ Wm,
    const float* __restrict__ av, const unsigned char* __restrict__ mask,
    float* __restrict__ out, int Btot)
{
    extern __shared__ __align__(128) char smem[];
    __bf16* sA   = (__bf16*)(smem + OFF_A);
    __bf16* sB   = (__bf16*)(smem + OFF_B);
    float*  sH   = (float*)(smem + OFF_H);
    float*  sLog = (float*)(smem + OFF_LOG);
    float*  sEs  = (float*)(smem + OFF_ES);
    float*  sWt  = (float*)(smem + OFF_W);
    float*  sRed = (float*)(smem + OFF_RED);
    float*  sPart= (float*)(smem + OFF_PART);
    float*  sAv  = (float*)(smem + OFF_AV);

    const int b   = blockIdx.x;
    const int tid = threadIdx.x;

    // ---- Stage h[b] -> bf16 A (rows 129..143 zero) ----
    const float* hb = h + (size_t)b * NROW * KD;
    for (int idx = tid; idx < MPAD * KD; idx += 256) {
        int r = idx >> 7, c = idx & 127;
        float v = (r < NROW) ? hb[r * KD + c] : 0.f;
        sA[r * KP + c] = f2bf(v);
    }
    // ---- Stage W^T stacked over t: sB[n][k] = W[t][k][c], n = t*64+c ----
    for (int idx = tid; idx < NCOL * KD; idx += 256) {
        int n = idx >> 7, k = idx & 127;
        int t = n >> 6, c = n & 63;
        sB[n * KP + k] = f2bf(Wm[((size_t)t * KD + k) * CC + c]);
    }
    for (int idx = tid; idx < 2 * CC * NT; idx += 256) sAv[idx] = av[idx];
    __syncthreads();

    // ---- WMMA GEMM: A(144x128) x B(128x192) -> H(144x192) f32 in LDS ----
    const int lane = tid & 31;
    const int wv   = tid >> 5;
    const int lrow = lane & 15;
    const int lhi  = lane >> 4;
    for (int tile = wv; tile < (MPAD / 16) * (NCOL / 16); tile += 8) {
        int mt = tile / (NCOL / 16);
        int nt = tile % (NCOL / 16);
        const __bf16* ap = sA + (mt * 16 + lrow) * KP;
        const __bf16* bp = sB + (nt * 16 + lrow) * KP;
        v8f acc = {};
        #pragma unroll
        for (int kk = 0; kk < KD / 32; ++kk) {
            int ks = kk * 32 + lhi * 16;
            v16bf a8 = *(const v16bf*)(ap + ks);
            v16bf b8 = *(const v16bf*)(bp + ks);
            acc = __builtin_amdgcn_wmma_f32_16x16x32_bf16(
                false, a8, false, b8, (short)0, acc, false, false);
        }
        int ncol = nt * 16 + lrow;
        #pragma unroll
        for (int v = 0; v < 8; ++v) {
            int m = mt * 16 + v + 8 * lhi;
            sH[m * HP + ncol] = acc[v];
        }
    }
    __syncthreads();

    // ---- e_self[j][i] = <H[0, j*64: ], a1[i]> ----
    if (tid < 9) {
        int j = tid / 3, i = tid % 3;
        float d = 0.f;
        for (int c = 0; c < CC; ++c) d += sH[j * CC + c] * sAv[i * 128 + c];
        sEs[tid] = d;
    }
    // ---- e_n[grp][i][n] = <H[row, i*64: ], a2[i]>, mask folded to -inf ----
    for (int idx = tid; idx < 2 * NCOL; idx += 256) {
        int grp = idx / NCOL, rr = idx % NCOL, i = rr >> 6, n = rr & 63;
        int row = 1 + grp * 64 + n;
        float d = 0.f;
        for (int c = 0; c < CC; ++c) d += sH[row * HP + i * CC + c] * sAv[i * 128 + 64 + c];
        unsigned char mk = mask[((size_t)i * Btot + b) * NROW + row];
        sWt[idx] = mk ? -INFINITY : d;
    }
    __syncthreads();

    // ---- logits[grp][j][i][n] ----
    for (int idx = tid; idx < 2 * 576; idx += 256) {
        int grp = idx / 576, r = idx % 576;
        int j = r / 192, rr = r % 192, i = rr >> 6;
        sLog[idx] = sEs[j * 3 + i] + sWt[grp * 192 + rr];
    }
    __syncthreads();

    // ---- softmax: max reduce per group (128 threads each) ----
    {
        int grp = tid >> 7, l = tid & 127;
        float m = -INFINITY;
        for (int r = l; r < 576; r += 128) m = fmaxf(m, sLog[grp * 576 + r]);
        sRed[tid] = m;
    }
    __syncthreads();
    for (int s = 64; s > 0; s >>= 1) {
        if ((tid & 127) < s) sRed[tid] = fmaxf(sRed[tid], sRed[tid + s]);
        __syncthreads();
    }
    if (tid == 0)   sEs[9]  = sRed[0];
    if (tid == 128) sEs[10] = sRed[128];
    __syncthreads();
    // ---- sum reduce ----
    {
        int grp = tid >> 7, l = tid & 127;
        float gm = sEs[9 + grp];
        float s = 0.f;
        for (int r = l; r < 576; r += 128) s += __expf(sLog[grp * 576 + r] - gm);
        sRed[tid] = s;
    }
    __syncthreads();
    for (int s = 64; s > 0; s >>= 1) {
        if ((tid & 127) < s) sRed[tid] += sRed[tid + s];
        __syncthreads();
    }
    if (tid == 0)   sEs[11] = sRed[0];
    if (tid == 128) sEs[12] = sRed[128];
    __syncthreads();

    // ---- weights summed over j: w[grp][i][n] ----
    for (int idx = tid; idx < 2 * NCOL; idx += 256) {
        int grp = idx / NCOL, rr = idx % NCOL;
        float gm = sEs[9 + grp], gs = sEs[11 + grp];
        float w = 0.f;
        #pragma unroll
        for (int j = 0; j < 3; ++j) w += __expf(sLog[grp * 576 + j * 192 + rr] - gm);
        sWt[idx] = w / gs;
    }
    __syncthreads();

    // ---- weighted aggregation, 4-way split over (grp,i,n), per channel c ----
    {
        int part = tid >> 6, c = tid & 63;
        float acc = 0.f;
        for (int idx = part * 96; idx < part * 96 + 96; ++idx) {
            int grp = idx / NCOL, rr = idx % NCOL, i = rr >> 6, n = rr & 63;
            int row = 1 + grp * 64 + n;
            acc += sWt[idx] * sH[row * HP + i * CC + c];
        }
        sPart[tid] = acc;
    }
    __syncthreads();
    if (tid < CC) {
        float r = sPart[tid] + sPart[64 + tid] + sPart[128 + tid] + sPart[192 + tid];
        #pragma unroll
        for (int t = 0; t < NT; ++t) {
            if (mask[((size_t)t * Btot + b) * NROW]) r += sH[t * CC + tid];
        }
        out[(size_t)b * CC + tid] = (r > 0.f) ? r : (__expf(r) - 1.f);
    }
}

extern "C" void kernel_launch(void* const* d_in, const int* in_sizes, int n_in,
                              void* d_out, int out_size, void* d_ws, size_t ws_size,
                              hipStream_t stream) {
    const float*         h    = (const float*)d_in[0];
    const float*         W    = (const float*)d_in[1];
    const float*         a    = (const float*)d_in[2];
    const unsigned char* mask = (const unsigned char*)d_in[3];
    float* out = (float*)d_out;
    const int Btot = in_sizes[0] / (NROW * KD);   // 2048
    hetgat_fused<<<Btot, 256, SMEM_BYTES, stream>>>(h, W, a, mask, out, Btot);
}
